// StyleConvBase3DVel_47742856462495
// MI455X (gfx1250) — matrix operand order repaired and checked by reading, not compiled, not run on hardware
//
#include <hip/hip_runtime.h>
#include <hip/hip_bf16.h>
#include <math.h>

typedef __attribute__((ext_vector_type(16))) __bf16 v16bf;
typedef __attribute__((ext_vector_type(8)))  float  v8f;

namespace {
constexpr int CIN = 64, COUT = 64, KIDN = 27;
constexpr int KTOT = CIN * KIDN;          // 1728 = GEMM K
constexpr int NKC  = KTOT / 32;           // 54 K-chunks of 32
constexpr int DIN  = 64, DOUT = 62;
constexpr int OD_T = 4, OH_T = 4, OW_T = 8;      // 128 output positions / WG
constexpr int SP_D = OD_T + 2, SP_H = OH_T + 2, SP_W = OW_T + 2;
constexpr int SP_N = SP_D * SP_H * SP_W;         // 360 input positions / tile
constexpr int ROW  = 40;                          // padded bf16 row (80B, 16B-mult)
constexpr int TOFF = SP_N * ROW;                  // per-tensor LDS elems (14400)
constexpr long long YSZ = 2LL * COUT * DOUT * DOUT * DOUT;   // 30,505,984 per output
}

// A-pack layout (bf16): [b][tensor(Wn=0,dW=1)][mt(4)][kc(54)][lane(32)][e(16)]
__device__ __forceinline__ int apack_base(int b, int tens, int mt, int kc) {
  return ((((b * 2 + tens) * 4 + mt) * NKC + kc) << 9);
}

// ---------------------------------------------------------------------------
// Kernel 1: style modulation + demodulation + JVP tangent of the weights,
// pre-swizzled into the CDNA5 WMMA 16-bit A-fragment layout (bf16).
// K ordering: k = kernel_offset*64 + cin  (each 32-chunk has one offset).
// ---------------------------------------------------------------------------
__global__ __launch_bounds__(256)
void modulate_pack_kernel(const float* __restrict__ s,
                          const float* __restrict__ sw,     // (Cin, 2)
                          const float* __restrict__ sb,     // (Cin,)
                          const float* __restrict__ weight, // (Cout,Cin,3,3,3)
                          __bf16* __restrict__ apack) {
  const int b    = blockIdx.x >> 6;
  const int cout = blockIdx.x & 63;
  const int tid  = threadIdx.x;
  const float s0 = s[b * 2 + 0], s1 = s[b * 2 + 1];

  float pss = 0.f, psd = 0.f;
  for (int i = tid; i < KTOT; i += 256) {
    int cin = i / KIDN, kid = i - cin * KIDN;
    float wv   = weight[(cout * CIN + cin) * KIDN + kid];
    float smod = s0 * sw[cin * 2 + 0] + s1 * sw[cin * 2 + 1] + sb[cin];
    float wm   = wv * smod;
    pss += wm * wm;
    psd += wm * (wv * sw[cin * 2 + 1]);   // ds_mod = style_weight[:,1]
  }
  __shared__ float rss[256], rsd[256];
  rss[tid] = pss; rsd[tid] = psd;
  __syncthreads();
  for (int off = 128; off > 0; off >>= 1) {
    if (tid < off) { rss[tid] += rss[tid + off]; rsd[tid] += rsd[tid + off]; }
    __syncthreads();
  }
  const float norm  = sqrtf(rss[0] + 1e-8f);
  const float inv   = 1.f / norm;
  const float dnorm = -rsd[0] * inv * inv * inv;

  const int mt = cout >> 4, lm = cout & 15;
  for (int i = tid; i < 2 * KTOT; i += 256) {
    int tens = (i >= KTOT);
    int j  = i - tens * KTOT;
    int kc = j >> 5, hh = (j >> 4) & 1, e = j & 15;
    // 16-bit A-matrix lane/element -> K mapping: contiguous k = 16*half + e (matches B)
    int kb = e + 16 * hh;
    int k  = kc * 32 + kb;
    int kid = k >> 6;        // CIN == 64
    int cin = k & 63;
    float wv   = weight[(cout * CIN + cin) * KIDN + kid];
    float smod = s0 * sw[cin * 2 + 0] + s1 * sw[cin * 2 + 1] + sb[cin];
    float dsm  = sw[cin * 2 + 1];
    float val  = tens ? (wv * dsm * inv + (wv * smod) * dnorm)   // dW_total
                      : (wv * smod * inv);                       // W_normalized
    apack[apack_base(b, tens, mt, kc) + ((lm + 16 * hh) << 4) + e] = (__bf16)val;
  }
}

// ---------------------------------------------------------------------------
// Kernel 2: fused triple implicit-GEMM 3D conv via bf16 WMMA (f32 accum).
// 8 waves = 2 M-groups x 4 N-groups; each wave: 2 M-tiles x 2 N-subtiles,
// 12 accumulators -> 12 WMMAs per K-chunk against 4 A + 4 B fragment loads.
// bf16 staged once into LDS; fully unrolled 27-offset loop folds all B
// addresses into ds_load_b128 immediate offsets.
// ---------------------------------------------------------------------------
__global__ __launch_bounds__(256)
void styleconv_wmma_kernel(const float* __restrict__ x,
                           const float* __restrict__ dx,
                           const float* __restrict__ bias,
                           const __bf16* __restrict__ apack,
                           float* __restrict__ out) {
  int bt = blockIdx.x;
  const int owt = bt & 7;
  const int oht = (bt >> 3) & 15;
  const int odt = (bt >> 7) & 15;
  const int b   = bt >> 11;
  const int od0 = odt * OD_T, oh0 = oht * OH_T, ow0 = owt * OW_T;

  const int tid  = threadIdx.x;
  const int lane = tid & 31;
  const int wid  = tid >> 5;
  const int mt2 = wid & 1;            // M-group: covers mt = 2*mt2 + {0,1}
  const int ng  = wid >> 1;           // N-group: 32 positions

  __shared__ __align__(16) __bf16 xs[2 * TOFF];   // 57,600 B: [tensor][sp][cin]

  v8f accY[2][2]  = {{{}, {}}, {{}, {}}};
  v8f accD1[2][2] = {{{}, {}}, {{}, {}}};
  v8f accD2[2][2] = {{{}, {}}, {{}, {}}};

  // fragment lane geometry: n = lane%16, half = lane/16
  const int ncol = lane & 15, hf = lane >> 4;
  const int pos0 = ng * 32 + ncol;                // N-subtile 0
  const int pos1 = pos0 + 16;                     // N-subtile 1
  const int owl0 = pos0 & 7, ohl0 = (pos0 >> 3) & 3, odl0 = pos0 >> 5;
  const int owl1 = pos1 & 7, ohl1 = (pos1 >> 3) & 3, odl1 = pos1 >> 5;
  // per-lane LDS base rows (kernel-offset displacement is linear & uniform)
  const __bf16* bl0 = xs + hf * 16 + ((odl0 * SP_H + ohl0) * SP_W + owl0) * ROW;
  const __bf16* bl1 = xs + hf * 16 + ((odl1 * SP_H + ohl1) * SP_W + owl1) * ROW;

  for (int ch = 0; ch < 2; ++ch) {                // cin halves of 32
    const int cin0 = ch * 32;
    __syncthreads();
    // Stage x/dx tiles as bf16 (converted once), cin-pairs packed per b32 store.
    for (int i = 0; i < 45; ++i) {                // 45*256 == 2*16*360
      int idx  = tid + (i << 8);
      int tens = idx / (16 * SP_N);
      int rem  = idx - tens * (16 * SP_N);
      int cl2  = rem / SP_N;
      int sp   = rem - cl2 * SP_N;
      int spd  = sp / (SP_H * SP_W);
      int r2   = sp - spd * (SP_H * SP_W);
      int sph  = r2 / SP_W;
      int spw  = r2 - sph * SP_W;
      int d = od0 + spd; d = d < DIN ? d : DIN - 1;   // clamp edge tiles
      int h = oh0 + sph; h = h < DIN ? h : DIN - 1;
      int w = ow0 + spw; w = w < DIN ? w : DIN - 1;
      const float* src = tens ? dx : x;
      size_t g = (((size_t)(b * CIN + cin0 + 2 * cl2) * DIN + d) * DIN + h) * DIN + w;
      float v0 = src[g];
      float v1 = src[g + (size_t)DIN * DIN * DIN];
      union { __bf16 h2[2]; unsigned u; } pk;
      pk.h2[0] = (__bf16)v0; pk.h2[1] = (__bf16)v1;
      *(unsigned*)&xs[tens * TOFF + sp * ROW + 2 * cl2] = pk.u;
    }
    __syncthreads();

    // L2-resident A stream for this cin-half
    __builtin_prefetch(apack + apack_base(b, 0, mt2 * 2, ch) + (lane << 4), 0, 1);
    __builtin_prefetch(apack + apack_base(b, 1, mt2 * 2, ch) + (lane << 4), 0, 1);

#pragma unroll
    for (int kid = 0; kid < KIDN; ++kid) {
      const int kd = kid / 9, r9 = kid - kd * 9;
      const int kh = r9 / 3,  kw = r9 - kh * 3;
      const int ko = (kd * (SP_H * SP_W) + kh * SP_W + kw) * ROW;  // constant
      const int kc = kid * 2 + ch;

      v16bf bx[2], bd[2];
      bx[0] = *(const v16bf*)(bl0 + ko);
      bx[1] = *(const v16bf*)(bl1 + ko);
      bd[0] = *(const v16bf*)(bl0 + ko + TOFF);
      bd[1] = *(const v16bf*)(bl1 + ko + TOFF);

#pragma unroll
      for (int ms = 0; ms < 2; ++ms) {
        const int mt = mt2 * 2 + ms;
        v16bf awn = *(const v16bf*)(apack + apack_base(b, 0, mt, kc) + (lane << 4));
        v16bf adw = *(const v16bf*)(apack + apack_base(b, 1, mt, kc) + (lane << 4));
#pragma unroll
        for (int ns = 0; ns < 2; ++ns) {
          accY[ms][ns]  = __builtin_amdgcn_wmma_f32_16x16x32_bf16(
              false, awn, false, bx[ns], (short)0, accY[ms][ns],  false, false);
          accD1[ms][ns] = __builtin_amdgcn_wmma_f32_16x16x32_bf16(
              false, adw, false, bx[ns], (short)0, accD1[ms][ns], false, false);
          accD2[ms][ns] = __builtin_amdgcn_wmma_f32_16x16x32_bf16(
              false, awn, false, bd[ns], (short)0, accD2[ms][ns], false, false);
        }
      }
    }
  }

  // C/D layout: lane n = lane%16; VGPR r holds m = r + 8*(lane/16)
  const int odl[2] = {odl0, odl1}, ohl[2] = {ohl0, ohl1}, owl[2] = {owl0, owl1};
#pragma unroll
  for (int ns = 0; ns < 2; ++ns) {
    const int od = od0 + odl[ns], oh = oh0 + ohl[ns], ow = ow0 + owl[ns];
    if (od < DOUT && oh < DOUT && ow < DOUT) {
#pragma unroll
      for (int ms = 0; ms < 2; ++ms) {
#pragma unroll
        for (int r = 0; r < 8; ++r) {
          int cout = (mt2 * 2 + ms) * 16 + hf * 8 + r;
          size_t o = (((size_t)(b * COUT + cout) * DOUT + od) * DOUT + oh) * DOUT + ow;
          out[o]       = accY[ms][ns][r] + bias[cout];
          out[YSZ + o] = accD1[ms][ns][r] + accD2[ms][ns][r];
        }
      }
    }
  }
}

extern "C" void kernel_launch(void* const* d_in, const int* in_sizes, int n_in,
                              void* d_out, int out_size, void* d_ws, size_t ws_size,
                              hipStream_t stream) {
  (void)in_sizes; (void)n_in; (void)out_size; (void)ws_size;
  const float* x   = (const float*)d_in[0];
  const float* s   = (const float*)d_in[1];
  const float* dxp = (const float*)d_in[2];
  const float* sw  = (const float*)d_in[3];
  const float* sb  = (const float*)d_in[4];
  const float* wt  = (const float*)d_in[5];
  const float* bs  = (const float*)d_in[6];
  __bf16* apack = (__bf16*)d_ws;   // 884,736 bytes

  modulate_pack_kernel<<<2 * COUT, 256, 0, stream>>>(s, sw, sb, wt, apack);

  const int tiles = 2 * 16 * 16 * 8;   // B x od-tiles x oh-tiles x ow-tiles
  styleconv_wmma_kernel<<<tiles, 256, 0, stream>>>(x, dxp, bs, apack, (float*)d_out);
}
